// SpatialSelfAttention_755914244770
// MI455X (gfx1250) — matrix operand order repaired and checked by reading, not compiled, not run on hardware
//
#include <hip/hip_runtime.h>

// ---------------- CDNA5 WMMA types & helpers ----------------
typedef __attribute__((ext_vector_type(16))) __bf16 v16bf;
typedef __attribute__((ext_vector_type(8)))  float  v8f;

#if defined(__has_builtin)
#  if __has_builtin(__builtin_amdgcn_sched_barrier)
#    define SCHED_FENCE() __builtin_amdgcn_sched_barrier(0)
#  endif
#endif
#ifndef SCHED_FENCE
#  define SCHED_FENCE()
#endif

union BF16x16 {
  v16bf v;
  uint4 q[2];
  __bf16 h[16];
};

// A-operand (16x32 bf16, lane=M, VGPRs hold K pairs with half-wave K shuffle):
// span points at row start of a 32-element (64-byte) K-chunk, row-major storage.
// lanes 0-15 take K pairs {0..7,16..23}; lanes 16-31 take {8..15,24..31}.
static __device__ __forceinline__ v16bf load_a(const __bf16* span, int half) {
  BF16x16 u;
  const char* p = (const char*)span;
  u.q[0] = *(const uint4*)(p + (half << 4));
  u.q[1] = *(const uint4*)(p + 32 + (half << 4));
  return u.v;
}

// B-operand (32x16 bf16, lane l holds column N=l%16: 16 contiguous bf16 = 32 bytes).
static __device__ __forceinline__ v16bf load_b(const __bf16* p) {
  BF16x16 u;
  u.q[0] = ((const uint4*)p)[0];
  u.q[1] = ((const uint4*)p)[1];
  return u.v;
}

static __device__ __forceinline__ v8f wmma_bf16(v16bf a, v16bf b, v8f c) {
  // 8 args: (neg_a, A, neg_b, B, c_mod, C, reuse_a, reuse_b)
  return __builtin_amdgcn_wmma_f32_16x16x32_bf16(false, a, false, b, (short)0, c,
                                                 false, false);
}

// ---------------- Kernel 1: qk = W*x + b, emit bf16 operand tensors ---------
// Ka[b][n][c] = qk            (A-operand rows for S^T: key rows)
// Qc[b][n][c] = qk * (1/N)    (B-operand for S^T: lane=query reads contig c)
// Xc[b][c][n] = x             (A-operand rows for O^T = V^T * P^T)
__global__ __launch_bounds__(256) void sa_prep(const float* __restrict__ x,
                                               const float* __restrict__ W,
                                               const float* __restrict__ bias,
                                               __bf16* __restrict__ Ka,
                                               __bf16* __restrict__ Qc,
                                               __bf16* __restrict__ Xc) {
  const int N = 4096;
  const int blk = blockIdx.x;
  const int b  = blk >> 6;
  const int n0 = (blk & 63) << 6;
  const int tid = threadIdx.x;

  __shared__ __align__(16) float xs[64][64];   // [c][n_local]
  __shared__ __align__(16) float Ws[64][64];   // [c_out][c_in]
  __shared__ float bs[64];

  for (int i = tid; i < 1024; i += 256)
    ((float4*)&Ws[0][0])[i] = ((const float4*)W)[i];
  if (tid < 64) bs[tid] = bias[tid];
  const float* xb = x + (size_t)b * 64 * 4096;
  for (int i = tid; i < 1024; i += 256) {
    int c = i >> 4, j4 = (i & 15) << 2;
    *(float4*)&xs[c][j4] = *(const float4*)(xb + (size_t)c * 4096 + n0 + j4);
  }
  __syncthreads();

  // Xc: straight bf16 cast of x, coalesced (thread t: row c=t>>2, 16 cols)
  {
    const int c  = tid >> 2;
    const int j0 = (tid & 3) << 4;
    BF16x16 u;
#pragma unroll
    for (int i = 0; i < 16; ++i) u.h[i] = (__bf16)xs[c][j0 + i];
    uint4* dst = (uint4*)(Xc + ((size_t)(b * 64 + c) * 4096 + n0 + j0));
    dst[0] = u.q[0]; dst[1] = u.q[1];
  }

  const int nl = tid >> 2;          // 0..63 local spatial
  const int c0 = (tid & 3) << 4;    // 0,16,32,48

  // qk[c0..c0+15][n] = W * x + b
  float acc[16];
#pragma unroll
  for (int i = 0; i < 16; ++i) acc[i] = bs[c0 + i];
  for (int k = 0; k < 64; ++k) {
    float xv = xs[k][nl];
#pragma unroll
    for (int i = 0; i < 16; ++i) acc[i] += Ws[c0 + i][k] * xv;
  }

  const float qscale = 1.0f / (float)N;
  BF16x16 uk, uq;
#pragma unroll
  for (int i = 0; i < 16; ++i) {
    uk.h[i] = (__bf16)acc[i];             // unscaled -> Ka rows
    uq.h[i] = (__bf16)(acc[i] * qscale);  // scaled   -> Qc cols
  }
  uint4* kdst = (uint4*)(Ka + ((size_t)(b * 4096 + n0 + nl) * 64 + c0));
  kdst[0] = uk.q[0]; kdst[1] = uk.q[1];
#pragma unroll
  for (int i = 0; i < 16; ++i)
    Qc[(size_t)(b * 4096 + n0 + nl) * 64 + c0 + i] = uq.h[i];
}

// ---------------- Kernel 2: fused flash attention (transposed S) ------------
// Grid: B * N/64 blocks, 128 threads (4 waves, wave32). Wave w owns 16 query
// COLUMNS qbase..qbase+15. S^T[key][query] puts the softmax axis (keys) on the
// in-lane M dimension: reductions are in-register trees + ONE half-wave
// exchange, and the query B-operands are loop-invariant.
__global__ __launch_bounds__(128) void sa_attn(const __bf16* __restrict__ Ka,
                                               const __bf16* __restrict__ Qc,
                                               const __bf16* __restrict__ Xc,
                                               float* __restrict__ out) {
  const int N = 4096, C = 64;
  const int blk = blockIdx.x;
  const int b  = blk >> 6;
  const int q0 = (blk & 63) << 6;
  const int tid  = threadIdx.x;
  const int wv   = tid >> 5;
  const int lid  = tid & 31;
  const int half = lid >> 4;
  const int l15  = lid & 15;

  __shared__ __align__(16) __bf16 ldsP[4][64 * 16];   // per-wave P^T strip

  const __bf16* Kab = Ka + (size_t)b * N * C;
  const __bf16* Qcb = Qc + (size_t)b * C * N;
  const __bf16* Xcb = Xc + (size_t)b * C * N;

  const int qbase = q0 + wv * 16;

  // Loop-invariant query B-operands: lane l holds query l%16, c 0-31 (resp. 32-63)
  const v16bf bq0 = load_a(Qcb + (size_t)(qbase + l15) * C, half);
  const v16bf bq1 = load_a(Qcb + (size_t)(qbase + l15) * C + 32, half);

  const v8f vzero = {0.f, 0.f, 0.f, 0.f, 0.f, 0.f, 0.f, 0.f};
  v8f accO[4];
#pragma unroll
  for (int u = 0; u < 4; ++u) accO[u] = vzero;
  float rm = -1e30f, rl = 0.f;   // per-lane (per-query) running stats

  for (int m0 = 0; m0 < N; m0 += 64) {
    if (m0 + 64 < N) {   // prefetch next key block (global_prefetch_b8)
      __builtin_prefetch((const char*)(Kab + (size_t)(m0 + 64 + l15) * C), 0, 0);
      __builtin_prefetch((const char*)(Xcb + (size_t)l15 * N + m0 + 64), 0, 0);
    }

    // ---- A-operands: key rows (for S^T) and V^T rows (for O^T)
    v16bf ak[8];
#pragma unroll
    for (int t = 0; t < 4; ++t) {
      const __bf16* krow = Kab + (size_t)(m0 + 16 * t + l15) * C;
      ak[2 * t]     = load_a(krow, half);        // c 0-31
      ak[2 * t + 1] = load_a(krow + 32, half);   // c 32-63
    }
    v16bf av[8];
#pragma unroll
    for (int u = 0; u < 4; ++u) {
      const __bf16* vrow = Xcb + (size_t)(16 * u + l15) * N + m0;
      av[2 * u]     = load_a(vrow, half);        // keys m0+0..31
      av[2 * u + 1] = load_a(vrow + 32, half);   // keys m0+32..63
    }
    SCHED_FENCE();   // keep loads above the WMMAs (pipelined partial waits)

    // ---- S^T tiles: [key 16t.. | query qbase..] = keys x queries
    v8f st[4];
#pragma unroll
    for (int t = 0; t < 4; ++t) st[t] = wmma_bf16(ak[2 * t], bq0, vzero);
#pragma unroll
    for (int t = 0; t < 4; ++t) st[t] = wmma_bf16(ak[2 * t + 1], bq1, st[t]);
    SCHED_FENCE();

    // ---- softmax over keys: in-lane tree + single half-wave exchange
    float mr[8];
#pragma unroll
    for (int r = 0; r < 8; ++r)
      mr[r] = fmaxf(fmaxf(st[0][r], st[1][r]), fmaxf(st[2][r], st[3][r]));
    float lm = fmaxf(fmaxf(fmaxf(mr[0], mr[1]), fmaxf(mr[2], mr[3])),
                     fmaxf(fmaxf(mr[4], mr[5]), fmaxf(mr[6], mr[7])));
    lm = fmaxf(lm, __shfl_xor(lm, 16, 32));   // partner half holds keys 8-15/24-31

    const float nm   = fmaxf(rm, lm);
    const float corr = __expf(rm - nm);
    rm = nm;

    // P^T = exp(S^T - m): per-lane row sums, stage bf16 P^T in LDS strip
    float pr[8];
#pragma unroll
    for (int r = 0; r < 8; ++r) pr[r] = 0.f;
#pragma unroll
    for (int t = 0; t < 4; ++t) {
#pragma unroll
      for (int r = 0; r < 8; ++r) {
        float p = __expf(st[t][r] - nm);
        pr[r] += p;
        // P^T strip layout: [query][key_local]: key = 16t + 8*half + r
        ldsP[wv][l15 * 64 + 16 * t + 8 * half + r] = (__bf16)p;
      }
    }
    float psum = ((pr[0] + pr[1]) + (pr[2] + pr[3])) +
                 ((pr[4] + pr[5]) + (pr[6] + pr[7]));
    psum += __shfl_xor(psum, 16, 32);

    rl = rl * corr + psum;
#pragma unroll
    for (int u = 0; u < 4; ++u) {
#pragma unroll
      for (int r = 0; r < 8; ++r) accO[u][r] *= corr;
    }

    // ---- P^T B-operands from LDS: lane l reads query row l%16, keys 32j..32j+31
    v16bf bp0 = load_a(&ldsP[wv][l15 * 64], half);         // keys m0+0..31
    v16bf bp1 = load_a(&ldsP[wv][l15 * 64 + 32], half);    // keys m0+32..63

    // ---- O^T += V^T * P^T  (back-to-back WMMAs, operands resident)
#pragma unroll
    for (int u = 0; u < 4; ++u) accO[u] = wmma_bf16(av[2 * u], bp0, accO[u]);
#pragma unroll
    for (int u = 0; u < 4; ++u) accO[u] = wmma_bf16(av[2 * u + 1], bp1, accO[u]);
  }

  // ---- epilogue: out[b][c][n] = O^T[c][query] / rowsum(query)
  const float invl = 1.0f / rl;
  const int n = qbase + l15;
#pragma unroll
  for (int u = 0; u < 4; ++u) {
#pragma unroll
    for (int r = 0; r < 8; ++r) {
      const int c = 16 * u + 8 * half + r;
      out[(size_t)(b * 64 + c) * 4096 + n] = accO[u][r] * invl;
    }
  }
}

// ---------------- launch ----------------------------------------------------
extern "C" void kernel_launch(void* const* d_in, const int* in_sizes, int n_in,
                              void* d_out, int out_size, void* d_ws, size_t ws_size,
                              hipStream_t stream) {
  const float* x    = (const float*)d_in[0];
  const float* W    = (const float*)d_in[1];
  const float* bias = (const float*)d_in[2];
  float* out = (float*)d_out;

  char* ws = (char*)d_ws;
  const size_t tbytes = (size_t)4 * 4096 * 64 * sizeof(__bf16);  // 2 MB each
  __bf16* Ka = (__bf16*)(ws);
  __bf16* Qc = (__bf16*)(ws + tbytes);
  __bf16* Xc = (__bf16*)(ws + 2 * tbytes);

  sa_prep<<<256, 256, 0, stream>>>(x, W, bias, Ka, Qc, Xc);
  sa_attn<<<256, 128, 0, stream>>>(Ka, Qc, Xc, out);
}
